// HungarianLoss_33449205301677
// MI455X (gfx1250) — matrix operand order repaired and checked by reading, not compiled, not run on hardware
//
#include <hip/hip_runtime.h>
#include <hip/hip_bf16.h>

typedef __attribute__((ext_vector_type(16))) _Float16 v16h;
typedef __attribute__((ext_vector_type(8)))  _Float16 v8h;
typedef __attribute__((ext_vector_type(8)))  float    v8f;

#define HW      262144      // 512*512
#define NB      8           // batch
#define NS      16          // prediction slots
#define NT      8           // GT masks
#define CPB     128         // workgroups (chunks) per batch
#define WGSIZE  256         // 8 wave32 per workgroup
#define WAVES   8
#define KSPAN_WG   (HW / CPB)          // 2048 pixels per workgroup
#define KSPAN_WAVE (KSPAN_WG / WAVES)  // 256 pixels per wave
#define KITERS     (KSPAN_WAVE / 32)   // 8 WMMA steps of K=32

// fast sigmoid: v_mul + v_exp_f32 + v_add + v_rcp_f32 (no IEEE div sequence)
__device__ __forceinline__ float sig(float x) {
    return __builtin_amdgcn_rcpf(
        1.0f + __builtin_amdgcn_exp2f(x * -1.44269504088896340736f));
}

// ---------------------------------------------------------------------------
// Kernel 0: zero global accumulators in workspace
// ---------------------------------------------------------------------------
__global__ void hung_init(float* ws, int n) {
    int i = blockIdx.x * blockDim.x + threadIdx.x;
    if (i < n) ws[i] = 0.0f;
}

// ---------------------------------------------------------------------------
// Kernel 1: streaming pass. Per batch accumulate (all via v_wmma_f32_16x16x32_f16)
//   G[16][16]   = sigmoid(P) * sigmoid(P)^T
//   I[16][16]   = sigmoid(P) * [T | ones | 0]^T   (col 0..7 = targets,
//                                                  col 8    = ones -> psum)
//   tsum[8]     (integer accumulate on targets)
// Each pred/target element is loaded from HBM exactly once; sigmoid computed
// exactly once; A fragment derived from B fragment via an LDS-staged tile.
// ---------------------------------------------------------------------------
__global__ void __launch_bounds__(WGSIZE)
hung_main(const float* __restrict__ preds, const int* __restrict__ targ,
          float* __restrict__ G, float* __restrict__ I, float* __restrict__ TS)
{
    __shared__ __align__(32) _Float16 tile[WAVES][NS][32];  // 8 KB f16 tiles
    __shared__ float Gl[NS * NS];
    __shared__ float Il[NS * NS];
    __shared__ float tsl[NT];

    const int tid = threadIdx.x;
    for (int i = tid; i < NS * NS; i += WGSIZE) { Gl[i] = 0.0f; Il[i] = 0.0f; }
    if (tid < NT) tsl[tid] = 0.0f;
    __syncthreads();

    const int b     = blockIdx.x / CPB;
    const int chunk = blockIdx.x % CPB;
    const int wave  = tid >> 5;
    const int lane  = tid & 31;
    const int l16   = lane & 15;
    const int hi    = lane >> 4;
    const int kbase = chunk * KSPAN_WG + wave * KSPAN_WAVE;

    const float* rowp = preds + ((size_t)b * NS + l16) * HW;  // slot row
    const int*   rowt = targ  + ((size_t)b * NT + l16) * HW;  // target row (l16<NT)

    v8f cg = {};   // Gram accumulator  (16x16 f32 D layout)
    v8f ci = {};   // cross accumulator (col 8 = psum)
    int tsacc = 0;

    for (int it = 0; it < KITERS; ++it) {
        const int k0 = kbase + it * 32;

        // ---- B-gram fragment (32x16): lane col N=l16,
        // lanes 0-15 hold K=k0..k0+15, lanes 16-31 hold K=k0+16..k0+31.
        // This is also exactly one full coverage of the 16x32 tile.
        const float* src = rowp + k0 + hi * 16;
        float bv[16];
        *(float4*)&bv[0]  = *(const float4*)(src);
        *(float4*)&bv[4]  = *(const float4*)(src + 4);
        *(float4*)&bv[8]  = *(const float4*)(src + 8);
        *(float4*)&bv[12] = *(const float4*)(src + 12);
        __builtin_prefetch(src + 32, 0, 1);   // next K step of this stream

        v16h bg;
        #pragma unroll
        for (int e = 0; e < 16; ++e) bg[e] = (_Float16)sig(bv[e]);

        // ---- stage f16 tile in LDS, re-read in A layout (16x32):
        // lane(l16,hi): elements 0..15 <- K=k0+hi*16.. (same order as B)
        // DS ops are in-order within a wave: no barrier needed.
        *(v16h*)&tile[wave][l16][hi * 16] = bg;                  // 2x ds_store_b128
        v8h alo = *(const v8h*)&tile[wave][l16][hi * 16];        // ds_load_b128
        v8h ahi = *(const v8h*)&tile[wave][l16][hi * 16 + 8];    // ds_load_b128
        v16h a  = __builtin_shufflevector(alo, ahi,
                    0, 1, 2, 3, 4, 5, 6, 7, 8, 9, 10, 11, 12, 13, 14, 15);

        // ---- B-cross fragment: cols 0..7 targets, col 8 ones (psum), rest 0
        v16h bt;
        if (l16 < NT) {
            const int* tsrc = rowt + k0 + hi * 16;
            int tv[16];
            *(int4*)&tv[0]  = *(const int4*)(tsrc);
            *(int4*)&tv[4]  = *(const int4*)(tsrc + 4);
            *(int4*)&tv[8]  = *(const int4*)(tsrc + 8);
            *(int4*)&tv[12] = *(const int4*)(tsrc + 12);
            #pragma unroll
            for (int e = 0; e < 16; ++e) {
                tsacc += tv[e];
                bt[e] = (_Float16)tv[e];
            }
        } else if (l16 == NT) {
            #pragma unroll
            for (int e = 0; e < 16; ++e) bt[e] = (_Float16)1.0f;
        } else {
            #pragma unroll
            for (int e = 0; e < 16; ++e) bt[e] = (_Float16)0.0f;
        }

        // full wave reconverged: EXEC all-1s for WMMA
        cg = __builtin_amdgcn_wmma_f32_16x16x32_f16(false, a, false, bg,
                                                    (short)0, cg, false, false);
        ci = __builtin_amdgcn_wmma_f32_16x16x32_f16(false, a, false, bt,
                                                    (short)0, ci, false, false);
    }

    if (l16 < NT) atomicAdd(&tsl[l16], (float)tsacc);

    // D layout: VGPR r holds (M = r + 8*hi, N = l16)
    #pragma unroll
    for (int r = 0; r < 8; ++r) {
        const int m = r + 8 * hi;
        atomicAdd(&Gl[m * NS + l16], cg[r]);
        atomicAdd(&Il[m * NS + l16], ci[r]);
    }
    __syncthreads();

    for (int i = tid; i < NS * NS; i += WGSIZE) {
        atomicAdd(&G[(size_t)b * NS * NS + i], Gl[i]);
        atomicAdd(&I[(size_t)b * NS * NS + i], Il[i]);
    }
    if (tid < NT) atomicAdd(&TS[b * NT + tid], tsl[tid]);
}

// ---------------------------------------------------------------------------
// Kernel 2: cost matrix + diversity reduce + per-batch Hungarian + scalar loss
// psum[b][s] lives in I[b][s][8] (the ones column of the cross WMMA).
// ---------------------------------------------------------------------------
__global__ void hung_finalize(const float* __restrict__ G, const float* __restrict__ I,
                              const float* __restrict__ TS, float* __restrict__ out)
{
    __shared__ float cost[NB][NS][NT];
    __shared__ float red[64];
    __shared__ float per_b[NB];
    const int tid = threadIdx.x;  // 64 threads

    // dice cost matrix (matched losses are exactly these entries)
    for (int idx = tid; idx < NB * NS * NT; idx += 64) {
        int b = idx / (NS * NT);
        int r = idx % (NS * NT);
        int s = r / NT, t = r % NT;
        float inter = I[b * NS * NS + s * NS + t];
        float ps = I[b * NS * NS + s * NS + NT];      // ones column = psum
        float ts = TS[b * NT + t];
        cost[b][s][t] = 1.0f - (2.0f * inter + 1.0f) / (ps + ts + 1.0f);
    }

    // diversity: off-diagonal sum of all Gram matrices
    float acc = 0.0f;
    for (int idx = tid; idx < NB * NS * NS; idx += 64) {
        int r = idx % (NS * NS);
        if ((r / NS) != (r % NS)) acc += G[idx];
    }
    red[tid] = acc;
    __syncthreads();
    for (int s = 32; s > 0; s >>= 1) {
        if (tid < s) red[tid] += red[tid + s];
        __syncthreads();
    }

    // Hungarian (Jonker-Volgenant potentials) per batch, transposed 8x16
    if (tid < NB) {
        const int b = tid;
        float u[NT + 1], v[NS + 1];
        int p[NS + 1], way[NS + 1];
        for (int i = 0; i <= NT; ++i) u[i] = 0.0f;
        for (int j = 0; j <= NS; ++j) { v[j] = 0.0f; p[j] = 0; way[j] = 0; }

        for (int i = 1; i <= NT; ++i) {
            p[0] = i;
            int j0 = 0;
            float minv[NS + 1];
            bool used[NS + 1];
            for (int j = 0; j <= NS; ++j) { minv[j] = 1e30f; used[j] = false; }
            for (;;) {
                used[j0] = true;
                const int i0 = p[j0];
                float delta = 1e30f;
                int j1 = 0;
                for (int j = 1; j <= NS; ++j) {
                    if (!used[j]) {
                        // transposed cost: row=target i0-1, col=slot j-1
                        float cur = cost[b][j - 1][i0 - 1] - u[i0] - v[j];
                        if (cur < minv[j]) { minv[j] = cur; way[j] = j0; }
                        if (minv[j] < delta) { delta = minv[j]; j1 = j; }
                    }
                }
                for (int j = 0; j <= NS; ++j) {
                    if (used[j]) { u[p[j]] += delta; v[j] -= delta; }
                    else         { minv[j] -= delta; }
                }
                j0 = j1;
                if (p[j0] == 0) break;
            }
            while (j0) { int j1 = way[j0]; p[j0] = p[j1]; j0 = j1; }
        }

        // p[j] != 0  =>  slot (j-1) matched to target (p[j]-1)
        float msum = 0.0f, fsum = 0.0f;
        for (int j = 1; j <= NS; ++j) {
            const int slot = j - 1;
            if (p[j] != 0) {
                msum += cost[b][slot][p[j] - 1];
            } else {
                float ps = I[b * NS * NS + slot * NS + NT];
                fsum += 1.0f - 1.0f / (ps + 1.0f);   // dice vs zero target
            }
        }
        per_b[b] = msum / (float)NT + 3.0f * (fsum / (float)(NS - NT));
    }
    __syncthreads();

    if (tid == 0) {
        float matching = 0.0f;
        for (int b = 0; b < NB; ++b) matching += per_b[b];
        matching /= (float)NB;
        float diversity = red[0] / ((float)HW * (float)NB);
        out[0] = matching + 1.0f * diversity;
    }
}

// ---------------------------------------------------------------------------
extern "C" void kernel_launch(void* const* d_in, const int* in_sizes, int n_in,
                              void* d_out, int out_size, void* d_ws, size_t ws_size,
                              hipStream_t stream) {
    const float* preds = (const float*)d_in[0];   // (8,16,512,512) f32
    const int*   targ  = (const int*)d_in[1];     // (8,8,512,512)  i32

    float* ws = (float*)d_ws;
    float* G  = ws;                 // NB*16*16 = 2048
    float* I  = ws + 2048;          // NB*16*16 = 2048
    float* TS = ws + 4096;          // NB*8     = 64
    const int accN = 4160;

    hung_init<<<(accN + 255) / 256, 256, 0, stream>>>(ws, accN);
    hung_main<<<NB * CPB, WGSIZE, 0, stream>>>(preds, targ, G, I, TS);
    hung_finalize<<<1, 64, 0, stream>>>(G, I, TS, (float*)d_out);
}
